// GroupedQueryAttention_58471684767877
// MI455X (gfx1250) — matrix operand order repaired and checked by reading, not compiled, not run on hardware
//
#include <hip/hip_runtime.h>

// ---------------------------------------------------------------------------
// GQA forward for MI455X (gfx1250, wave32, WMMA 16x16x32 bf16, TDM staging)
//   x[B,S,H] -> Q,K,V proj (bf16 WMMA GEMM) -> flash attention (WMMA) ->
//   O proj (bf16 WMMA GEMM, fp32 out)
// Non-transposed LDS tiles (GEMM A, attention Q/K) are staged with the CDNA5
// Tensor Data Mover (tensor_load_to_lds + s_wait_tensorcnt) when the builtin
// exists; transposed tiles (GEMM B, attention V^T) are staged manually.
// ---------------------------------------------------------------------------

typedef unsigned short u16;
typedef __bf16 v16bf __attribute__((ext_vector_type(16)));
typedef float  v8f   __attribute__((ext_vector_type(8)));
typedef unsigned int u32x4 __attribute__((ext_vector_type(4)));
typedef int i32x4 __attribute__((ext_vector_type(4)));
typedef int i32x8 __attribute__((ext_vector_type(8)));

#define DEV static __device__ __forceinline__

#if defined(__has_builtin)
#if __has_builtin(__builtin_amdgcn_tensor_load_to_lds) && \
    __has_builtin(__builtin_amdgcn_s_wait_tensorcnt)
#define HAVE_TDM 1
#endif
#endif

constexpr int HIDDEN = 2048;
constexpr int NUM_HEADS = 32;
constexpr int NUM_KV_HEADS = 8;
constexpr int HEAD_DIM = 64;
constexpr int KVSZ = 512;
constexpr int BATCH = 2;
constexpr int SEQ = 2048;
constexpr int BS = BATCH * SEQ;         // 4096 rows

// ---- bf16 helpers ----------------------------------------------------------
DEV u16 f2bf(float f) {
  unsigned u = __float_as_uint(f);
  u = (u + 0x7FFFu + ((u >> 16) & 1u)) >> 16;   // round-to-nearest-even
  return (u16)u;
}

struct U4x2 { uint4 lo; uint4 hi; };

// Build a 16-element bf16 fragment from two 16-byte LDS reads.
DEV v16bf ld_frag(const u16* p0, const u16* p1) {
  U4x2 u;
  u.lo = *reinterpret_cast<const uint4*>(p0);
  u.hi = *reinterpret_cast<const uint4*>(p1);
  return __builtin_bit_cast(v16bf, u);
}

DEV v8f wmma_bf16(v16bf a, v16bf b, v8f c) {
  return __builtin_amdgcn_wmma_f32_16x16x32_bf16(
      /*neg_a=*/false, a, /*neg_b=*/false, b,
      /*c_mod=*/(short)0, c, /*reuse_a=*/false, /*reuse_b=*/false);
}

DEV v8f vzero8() {
  v8f z;
  #pragma unroll
  for (int i = 0; i < 8; ++i) z[i] = 0.0f;
  return z;
}

#ifdef HAVE_TDM
// ---- Tensor Data Mover: 2D bf16 tile (tile_h rows x tile_w elems) ----------
// global tile start `gaddr` with row stride `stride0` elems; LDS destination
// byte offset `lds_off`, with `pad_amount_dw` DWORDs of LDS padding inserted
// every 2^(pad_code+1) DWORDs (0 disables). Descriptor per ISA 08 S8.
DEV void tdm_load_2d_bf16(const void* gaddr, unsigned lds_off,
                          unsigned tile_w, unsigned tile_h, unsigned stride0,
                          unsigned pad_code, unsigned pad_amount_dw) {
  unsigned long long ga = (unsigned long long)(size_t)gaddr;
  u32x4 g0;
  g0[0] = 1u;                                   // count=1, user mode, no gather
  g0[1] = lds_off;                              // lds_addr
  g0[2] = (unsigned)(ga & 0xFFFFFFFFu);         // global_addr[31:0]
  g0[3] = (unsigned)((ga >> 32) & 0x01FFFFFFu)  // global_addr[56:32]
          | (2u << 30);                         // type = 2 ("image")
  i32x8 g1;
  unsigned dw0 = (1u << 16);                    // data_size=1 (2B), mask=0
  if (pad_amount_dw)
    dw0 |= (1u << 20) | (pad_code << 22) | ((pad_amount_dw - 1u) << 25);
  g1[0] = (int)dw0;
  g1[1] = (int)((tile_w & 0xFFFFu) << 16);      // tensor_dim0[15:0] (=tile_w)
  g1[2] = (int)((tile_w >> 16) | ((tile_h & 0xFFFFu) << 16));  // tensor_dim1
  g1[3] = (int)((tile_h >> 16) | (tile_w << 16));              // tile_dim0
  g1[4] = (int)tile_h;                          // tile_dim1 (tile_dim2 = 0)
  g1[5] = (int)stride0;                         // tensor_dim0_stride[31:0]
  g1[6] = 0;                                    // stride hi / dim1_stride lo
  g1[7] = 0;
  i32x4 g2; g2[0] = 1; g2[1] = 0; g2[2] = 0; g2[3] = 0;   // tensor_dim2=1
  i32x4 g3; g3[0] = 0; g3[1] = 0; g3[2] = 0; g3[3] = 0;
#if __clang_major__ >= 23
  i32x8 g4;
  #pragma unroll
  for (int i = 0; i < 8; ++i) g4[i] = 0;
  __builtin_amdgcn_tensor_load_to_lds(g0, g1, g2, g3, g4, 0);
#else
  __builtin_amdgcn_tensor_load_to_lds(g0, g1, g2, g3, 0);
#endif
}
#endif  // HAVE_TDM

// ---- fp32 -> bf16 conversion ----------------------------------------------
__global__ void __launch_bounds__(256)
cvt_f32_bf16(const float* __restrict__ in, u16* __restrict__ out, int n) {
  int i = (blockIdx.x * blockDim.x + threadIdx.x) * 4;
  if (i >= n) return;
  float4 f = *reinterpret_cast<const float4*>(in + i);
  ushort4 o;
  o.x = f2bf(f.x); o.y = f2bf(f.y); o.z = f2bf(f.z); o.w = f2bf(f.w);
  *reinterpret_cast<ushort4*>(out + i) = o;
}

// ---- generic bf16 GEMM: C[M,N] = A[M,K]*B[K,N] + bias ----------------------
// 256 threads = 8 waves; block tile 128x128, k-step 32.
// Wave grid 4x2; each wave owns 32x64 = 2x4 fragments of 16x16.
template <bool OUT_BF16>
__global__ void __launch_bounds__(256)
gemm_bf16(const u16* __restrict__ A, const u16* __restrict__ B,
          const float* __restrict__ bias, void* __restrict__ Cout,
          int M, int N, int K) {
  constexpr int BM = 128, BN = 128, BK = 32;
  constexpr int AST = BK + 8;   // padded LDS stride (bytes multiple of 16)
  constexpr int BST = BK + 8;
  __shared__ u16 As[BM * AST];          // [row][k]
  __shared__ u16 Bs[BN * BST];          // transposed: [n][k]

  const int bm = blockIdx.y * BM, bn = blockIdx.x * BN;
  const int t = threadIdx.x;
  const int lane = t & 31, w = t >> 5;
  const int wm = (w & 3) * 32, wn = (w >> 2) * 64;
  const int half = lane >> 4, lr = lane & 15;

  v8f acc[2][4];
  #pragma unroll
  for (int fr = 0; fr < 2; ++fr)
    #pragma unroll
    for (int fc = 0; fc < 4; ++fc) acc[fr][fc] = vzero8();

  for (int k0 = 0; k0 < K; k0 += BK) {
    __syncthreads();
#ifdef HAVE_TDM
    // A tile 128x32 via Tensor DMA (one descriptor, wave 0 issues).
    // LDS row = 32 elems = 16 DW, then 4 DW pad  ->  stride 40 elems.
    if (w == 0)
      tdm_load_2d_bf16(A + (size_t)bm * K + k0, (unsigned)(size_t)(void*)As,
                       /*tile_w=*/BK, /*tile_h=*/BM, /*stride0=*/(unsigned)K,
                       /*pad_code(16DW)=*/3, /*pad_dw=*/4);
#else
    // stage A tile manually: each thread 16 contiguous bf16 (2 x b128)
    {
      int r = t >> 1, c0 = (t & 1) * 16;
      const u16* g = A + (size_t)(bm + r) * K + k0 + c0;
      *reinterpret_cast<uint4*>(&As[r * AST + c0])     = *reinterpret_cast<const uint4*>(g);
      *reinterpret_cast<uint4*>(&As[r * AST + c0 + 8]) = *reinterpret_cast<const uint4*>(g + 8);
    }
#endif
    // stage B tile transposed: read 16 contiguous along n, scatter to [n][k]
    {
      int c = t & 31, n0 = (t >> 5) * 16;
      const u16* g = B + (size_t)(k0 + c) * N + bn + n0;
      u16 tmp[16];
      *reinterpret_cast<uint4*>(&tmp[0]) = *reinterpret_cast<const uint4*>(g);
      *reinterpret_cast<uint4*>(&tmp[8]) = *reinterpret_cast<const uint4*>(g + 8);
      #pragma unroll
      for (int i = 0; i < 16; ++i) Bs[(n0 + i) * BST + c] = tmp[i];
    }
    // prefetch next B tile (global_prefetch_b8)
    if (k0 + BK < K)
      __builtin_prefetch(B + (size_t)(k0 + BK + (t & 31)) * N + bn, 0, 1);
#ifdef HAVE_TDM
    if (w == 0) __builtin_amdgcn_s_wait_tensorcnt(0);
#endif
    __syncthreads();

    // hoist the 4 B fragments; reuse across both A fragment rows
    v16bf bfr[4];
    #pragma unroll
    for (int fc = 0; fc < 4; ++fc) {
      const u16* bp = &Bs[(wn + fc * 16 + lr) * BST + half * 16];
      bfr[fc] = ld_frag(bp, bp + 8);
    }
    #pragma unroll
    for (int fr = 0; fr < 2; ++fr) {
      const u16* ap = &As[(wm + fr * 16 + lr) * AST + half * 16];
      v16bf a = ld_frag(ap, ap + 8);
      #pragma unroll
      for (int fc = 0; fc < 4; ++fc)
        acc[fr][fc] = wmma_bf16(a, bfr[fc], acc[fr][fc]);
    }
  }

  // epilogue: D element i -> row = i + 8*half, col = lane%16 (per ISA layout)
  #pragma unroll
  for (int fr = 0; fr < 2; ++fr) {
    #pragma unroll
    for (int fc = 0; fc < 4; ++fc) {
      int col = bn + wn + fc * 16 + lr;
      float bv = bias[col];
      #pragma unroll
      for (int i = 0; i < 8; ++i) {
        int row = bm + wm + fr * 16 + half * 8 + i;
        float v = acc[fr][fc][i] + bv;
        if (OUT_BF16)
          reinterpret_cast<u16*>(Cout)[(size_t)row * N + col] = f2bf(v);
        else
          reinterpret_cast<float*>(Cout)[(size_t)row * N + col] = v;
      }
    }
  }
}

// ---- flash attention: one (head, 64-row q block) per workgroup -------------
// 128 threads = 4 waves; wave w owns q rows [w*16, w*16+16).
__global__ void __launch_bounds__(128)
attn_kernel(const u16* __restrict__ Q, const u16* __restrict__ Kx,
            const u16* __restrict__ Vx, const float* __restrict__ mask,
            u16* __restrict__ ctx) {
  constexpr int ST = 72;                       // padded stride, 144B (mult 16)
  __shared__ u16 Qs[64 * ST];                  // [q row][d]
  __shared__ u16 Ks[64 * ST];                  // [key][d]
  __shared__ u16 Vt[64 * ST];                  // transposed: [d][key]
  __shared__ u16 Ps[4][16 * ST];               // per-wave P staging [m][key]
  __shared__ float maskS[64];

  const int qb = blockIdx.x, head = blockIdx.y;
  const int b = head >> 5, hq = head & 31, kv = hq >> 2;
  const int t = threadIdx.x;
  const int lane = t & 31, w = t >> 5;
  const int half = lane >> 4, lr = lane & 15;

  // stage Q tile (64x64) once; LDS row = 64 elems = 32 DW, pad 4 DW -> 72
#ifdef HAVE_TDM
  if (w == 0)
    tdm_load_2d_bf16(Q + (size_t)(b * SEQ + qb * 64) * HIDDEN + hq * HEAD_DIM,
                     (unsigned)(size_t)(void*)Qs,
                     /*tile_w=*/64, /*tile_h=*/64, /*stride0=*/HIDDEN,
                     /*pad_code(32DW)=*/4, /*pad_dw=*/4);
#else
  {
    int r = t >> 1, c0 = (t & 1) * 32;
    const u16* g = Q + (size_t)(b * SEQ + qb * 64 + r) * HIDDEN + hq * HEAD_DIM + c0;
    #pragma unroll
    for (int i = 0; i < 4; ++i)
      *reinterpret_cast<uint4*>(&Qs[r * ST + c0 + i * 8]) =
          *reinterpret_cast<const uint4*>(g + i * 8);
  }
#endif

  float m_prev[8], lsum[8];
  v8f acc_o[4];
  #pragma unroll
  for (int i = 0; i < 8; ++i) { m_prev[i] = -1e30f; lsum[i] = 0.0f; }
  #pragma unroll
  for (int fc = 0; fc < 4; ++fc) acc_o[fc] = vzero8();

  for (int kb = 0; kb < SEQ / 64; ++kb) {
    __syncthreads();   // previous iteration's LDS consumers are done
    // stage K tile [key][d]
#ifdef HAVE_TDM
    if (w == 0)
      tdm_load_2d_bf16(Kx + (size_t)(b * SEQ + kb * 64) * KVSZ + kv * HEAD_DIM,
                       (unsigned)(size_t)(void*)Ks,
                       /*tile_w=*/64, /*tile_h=*/64, /*stride0=*/KVSZ,
                       /*pad_code(32DW)=*/4, /*pad_dw=*/4);
#else
    {
      int r = t >> 1, c0 = (t & 1) * 32;
      const u16* g = Kx + (size_t)(b * SEQ + kb * 64 + r) * KVSZ + kv * HEAD_DIM + c0;
      #pragma unroll
      for (int i = 0; i < 4; ++i)
        *reinterpret_cast<uint4*>(&Ks[r * ST + c0 + i * 8]) =
            *reinterpret_cast<const uint4*>(g + i * 8);
    }
#endif
    // stage V tile transposed [d][key] (TDM cannot transpose -> manual)
    {
      int key = t >> 1, c0 = (t & 1) * 32;
      const u16* g = Vx + (size_t)(b * SEQ + kb * 64 + key) * KVSZ + kv * HEAD_DIM + c0;
      u16 tmp[32];
      #pragma unroll
      for (int i = 0; i < 4; ++i)
        *reinterpret_cast<uint4*>(&tmp[i * 8]) = *reinterpret_cast<const uint4*>(g + i * 8);
      #pragma unroll
      for (int i = 0; i < 32; ++i) Vt[(c0 + i) * ST + key] = tmp[i];
    }
    if (t < 64) maskS[t] = mask[b * SEQ + kb * 64 + t];
#ifdef HAVE_TDM
    if (w == 0) __builtin_amdgcn_s_wait_tensorcnt(0);   // covers Q (kb==0) + K
#endif
    __syncthreads();

    // S = (Q K^T)*scale + mask : hoist A per k-half, 4 key-tiles each
    v8f sacc[4];
    #pragma unroll
    for (int fc = 0; fc < 4; ++fc) sacc[fc] = vzero8();
    #pragma unroll
    for (int ki = 0; ki < 64; ki += 32) {
      const u16* ap = &Qs[(w * 16 + lr) * ST + ki + half * 16];
      v16bf a = ld_frag(ap, ap + 8);
      #pragma unroll
      for (int fc = 0; fc < 4; ++fc) {
        const u16* bp = &Ks[(fc * 16 + lr) * ST + ki + half * 16];
        v16bf bm = ld_frag(bp, bp + 8);
        sacc[fc] = wmma_bf16(a, bm, sacc[fc]);
      }
    }
    float p[4][8];
    #pragma unroll
    for (int fc = 0; fc < 4; ++fc) {
      float mk = maskS[fc * 16 + lr];
      #pragma unroll
      for (int i = 0; i < 8; ++i) p[fc][i] = sacc[fc][i] * 0.125f + mk;
    }

    // online softmax over this key block (rows i+8*half, reduce across 16 lanes)
    float rm[8];
    #pragma unroll
    for (int i = 0; i < 8; ++i) {
      rm[i] = fmaxf(fmaxf(p[0][i], p[1][i]), fmaxf(p[2][i], p[3][i]));
      rm[i] = fmaxf(rm[i], __shfl_xor(rm[i], 1));
      rm[i] = fmaxf(rm[i], __shfl_xor(rm[i], 2));
      rm[i] = fmaxf(rm[i], __shfl_xor(rm[i], 4));
      rm[i] = fmaxf(rm[i], __shfl_xor(rm[i], 8));
    }
    float sc[8];
    #pragma unroll
    for (int i = 0; i < 8; ++i) {
      float mn = fmaxf(m_prev[i], rm[i]);
      sc[i] = __expf(m_prev[i] - mn);
      m_prev[i] = mn;
      lsum[i] *= sc[i];
    }
    #pragma unroll
    for (int fc = 0; fc < 4; ++fc)
      #pragma unroll
      for (int i = 0; i < 8; ++i) p[fc][i] = __expf(p[fc][i] - m_prev[i]);
    float rs[8];
    #pragma unroll
    for (int i = 0; i < 8; ++i) {
      rs[i] = p[0][i] + p[1][i] + p[2][i] + p[3][i];
      rs[i] += __shfl_xor(rs[i], 1);
      rs[i] += __shfl_xor(rs[i], 2);
      rs[i] += __shfl_xor(rs[i], 4);
      rs[i] += __shfl_xor(rs[i], 8);
      lsum[i] += rs[i];
    }
    #pragma unroll
    for (int fc = 0; fc < 4; ++fc)
      #pragma unroll
      for (int i = 0; i < 8; ++i) acc_o[fc][i] *= sc[i];

    // restage P (D-layout -> A-layout) through per-wave LDS
    #pragma unroll
    for (int fc = 0; fc < 4; ++fc)
      #pragma unroll
      for (int i = 0; i < 8; ++i)
        Ps[w][(half * 8 + i) * ST + fc * 16 + lr] = f2bf(p[fc][i]);
    __syncthreads();

    // O += P * V : hoist A per key-half, 4 d-tiles each
    #pragma unroll
    for (int ki = 0; ki < 64; ki += 32) {
      const u16* ap = &Ps[w][lr * ST + ki + half * 16];
      v16bf a = ld_frag(ap, ap + 8);
      #pragma unroll
      for (int fc = 0; fc < 4; ++fc) {
        const u16* bp = &Vt[(fc * 16 + lr) * ST + ki + half * 16];
        v16bf bm = ld_frag(bp, bp + 8);
        acc_o[fc] = wmma_bf16(a, bm, acc_o[fc]);
      }
    }
  }

  // write context (bf16) at [b, s, head*64 + d]
  #pragma unroll
  for (int fc = 0; fc < 4; ++fc) {
    int col = hq * HEAD_DIM + fc * 16 + lr;
    #pragma unroll
    for (int i = 0; i < 8; ++i) {
      int row = b * SEQ + qb * 64 + w * 16 + half * 8 + i;
      ctx[(size_t)row * HIDDEN + col] = f2bf(acc_o[fc][i] / lsum[i]);
    }
  }
}

// ---------------------------------------------------------------------------
extern "C" void kernel_launch(void* const* d_in, const int* in_sizes, int n_in,
                              void* d_out, int out_size, void* d_ws, size_t ws_size,
                              hipStream_t stream) {
  const float* x    = (const float*)d_in[0];
  const float* mask = (const float*)d_in[1];
  const float* Wq   = (const float*)d_in[2];
  const float* bq   = (const float*)d_in[3];
  const float* Wk   = (const float*)d_in[4];
  const float* bk   = (const float*)d_in[5];
  const float* Wv   = (const float*)d_in[6];
  const float* bv   = (const float*)d_in[7];
  const float* Wo   = (const float*)d_in[8];
  const float* bo   = (const float*)d_in[9];
  float* out = (float*)d_out;

  // workspace carve-out (bf16 buffers), ~76 MB total
  char* p = (char*)d_ws;
  u16* xb  = (u16*)p; p += (size_t)BS * HIDDEN * 2;
  u16* wqb = (u16*)p; p += (size_t)HIDDEN * HIDDEN * 2;
  u16* wkb = (u16*)p; p += (size_t)HIDDEN * KVSZ * 2;
  u16* wvb = (u16*)p; p += (size_t)HIDDEN * KVSZ * 2;
  u16* wob = (u16*)p; p += (size_t)HIDDEN * HIDDEN * 2;
  u16* qb  = (u16*)p; p += (size_t)BS * HIDDEN * 2;
  u16* kb  = (u16*)p; p += (size_t)BS * KVSZ * 2;
  u16* vb  = (u16*)p; p += (size_t)BS * KVSZ * 2;
  u16* ctx = (u16*)p; p += (size_t)BS * HIDDEN * 2;

  auto cvt = [&](const float* src, u16* dst, int n) {
    cvt_f32_bf16<<<dim3((n / 4 + 255) / 256), 256, 0, stream>>>(src, dst, n);
  };
  cvt(x,  xb,  BS * HIDDEN);
  cvt(Wq, wqb, HIDDEN * HIDDEN);
  cvt(Wk, wkb, HIDDEN * KVSZ);
  cvt(Wv, wvb, HIDDEN * KVSZ);
  cvt(Wo, wob, HIDDEN * HIDDEN);

  // Q/K/V projections (bf16 out)
  gemm_bf16<true><<<dim3(HIDDEN / 128, BS / 128), 256, 0, stream>>>(
      xb, wqb, bq, qb, BS, HIDDEN, HIDDEN);
  gemm_bf16<true><<<dim3(KVSZ / 128, BS / 128), 256, 0, stream>>>(
      xb, wkb, bk, kb, BS, KVSZ, HIDDEN);
  gemm_bf16<true><<<dim3(KVSZ / 128, BS / 128), 256, 0, stream>>>(
      xb, wvb, bv, vb, BS, KVSZ, HIDDEN);

  // flash attention: grid (q-blocks, batch*heads)
  attn_kernel<<<dim3(SEQ / 64, BATCH * NUM_HEADS), 128, 0, stream>>>(
      qb, kb, vb, mask, ctx);

  // output projection (fp32 out)
  gemm_bf16<false><<<dim3(HIDDEN / 128, BS / 128), 256, 0, stream>>>(
      ctx, wob, bo, out, BS, HIDDEN, HIDDEN);
}
